// KD_EGNN_edge_61993557950951
// MI455X (gfx1250) — matrix operand order repaired and checked by reading, not compiled, not run on hardware
//
#include <hip/hip_runtime.h>
#include <math.h>

// ---------------------------------------------------------------------------
// EGNN pipeline for MI455X (gfx1250).  Dense GEMMs on the WMMA pipe
// (v_wmma_f32_16x16x32_bf16, f32 accumulate).  Weights are pre-packed once
// per launch into bf16-pair, column-major tiles so B fragments are direct
// global b128 loads (L2-resident); activations are staged fp32->bf16 via LDS
// with software pipelining.  Gather/scatter on VALU + global f32 atomics.
// ---------------------------------------------------------------------------

typedef __attribute__((ext_vector_type(16))) __bf16 bf16x16;
typedef __attribute__((ext_vector_type(2)))  __bf16 bf16x2;
typedef __attribute__((ext_vector_type(8)))  float  f32x8;
typedef __attribute__((ext_vector_type(2)))  float  f32x2;

union Frag { bf16x16 v; unsigned int u[8]; uint4 q[2]; };

enum { ACT_NONE = 0, ACT_SILU = 1, ACT_RELU = 2, ACT_SIGMOID = 3, ACT_TANH = 4 };

__device__ __forceinline__ float act_apply(float v, int act) {
  switch (act) {
    case ACT_SILU:    return v / (1.0f + __expf(-v));
    case ACT_RELU:    return v > 0.0f ? v : 0.0f;
    case ACT_SIGMOID: return 1.0f / (1.0f + __expf(-v));
    case ACT_TANH:    return tanhf(v);
    default:          return v;
  }
}

// packed bf16 pair: even k -> [15:0], odd k -> [31:16]  (single v_cvt_pk_bf16_f32)
__device__ __forceinline__ unsigned int pack2bf(float a, float b) {
  f32x2 s; s.x = a; s.y = b;
  union { bf16x2 v; unsigned int u; } x;
  x.v = __builtin_convertvector(s, bf16x2);
  return x.u;
}

// Stage one 32-deep K-slab of the activation matrix A (8 floats along K).
__device__ __forceinline__ void loadA(const float* __restrict__ Arow,
                                      int k0, int K, int ka, float ar[8])
{
  if (k0 + 32 <= K) {
    const float4* pa = (const float4*)(Arow + k0 + ka);
    float4 a0 = pa[0], a1 = pa[1];
    ar[0] = a0.x; ar[1] = a0.y; ar[2] = a0.z; ar[3] = a0.w;
    ar[4] = a1.x; ar[5] = a1.y; ar[6] = a1.z; ar[7] = a1.w;
  } else {
#pragma unroll
    for (int i = 0; i < 8; ++i) {
      int gk = k0 + ka + i;
      float v = Arow[min(gk, K - 1)];
      ar[i] = (gk < K) ? v : 0.0f;
    }
  }
}

// ---------------------------------------------------------------------------
// Pack weight W[K,N] (fp32, row-major) -> Wt[n][kp2] of uint bf16-pairs,
// K zero-padded to kp2*2 (kp2 = 16 * ceil(K/32)).
// ---------------------------------------------------------------------------
__global__ void pack_weight_kernel(const float* __restrict__ W, unsigned int* __restrict__ Wt,
                                   int K, int N, int kp2)
{
  long long idx = (long long)blockIdx.x * blockDim.x + threadIdx.x;
  if (idx >= (long long)N * kp2) return;
  int n  = (int)(idx / kp2);
  int kp = (int)(idx % kp2);
  int k0 = kp * 2, k1 = k0 + 1;
  float a = (k0 < K) ? W[(long long)k0 * N + n] : 0.0f;
  float b = (k1 < K) ? W[(long long)k1 * N + n] : 0.0f;
  Wt[idx] = pack2bf(a, b);
}

// ---------------------------------------------------------------------------
// WMMA GEMM:  C[M,N] = act(A[M,K] @ W[K,N] + bias) (+ resid)
// A fp32 row stride lda (lda % 4 == 0); Wt packed bf16 pairs [N][kp2].
// Requires N % 128 == 0.  Block: 256 threads = 8 waves (2x4), tile 64x128,
// BK=32; wave computes 32x32 via 2x2 v_wmma_f32_16x16x32_bf16.
// B fragments: two global_load_b128 per frag, exact WMMA register image.
// ---------------------------------------------------------------------------
__global__ __launch_bounds__(256)
void gemm_bf16_wmma_kernel(const float* __restrict__ A, const unsigned int* __restrict__ Wt,
                           const float* __restrict__ bias, const float* __restrict__ resid,
                           float* __restrict__ C, int M, int N, int K, int lda, int kp2, int act)
{
  __shared__ unsigned int lA[64 * 20];    // [row][kpair], stride 20 -> b128-aligned frag reads

  const int tid  = threadIdx.x;
  const int bm   = blockIdx.y * 64;
  const int bn   = blockIdx.x * 128;
  const int wave = tid >> 5;
  const int lane = tid & 31;
  const int wm   = wave >> 2;     // 0..1
  const int wn   = wave & 3;      // 0..3
  const int lrow = lane & 15;
  const int lhalf = lane >> 4;

  // A staging role: row ra, 8-float K-segment at ka.  Row clamp: out-of-range
  // A rows stage stale data whose C rows are never stored.
  const int ra = tid >> 2;
  const int ka = (tid & 3) * 8;
  const float* Arow = A + (long long)min(bm + ra, M - 1) * lda;

  // Per-lane packed-weight row pointers for the two B sub-tiles (lhalf offset
  // gives contiguous k order: kp = lhalf*8 + v, k = 16*lhalf + s).
  const unsigned int* wrow0 = Wt + (long long)(bn + wn * 32 +  0 + lrow) * kp2 + lhalf * 8;
  const unsigned int* wrow1 = Wt + (long long)(bn + wn * 32 + 16 + lrow) * kp2 + lhalf * 8;

  float ar[8];
  loadA(Arow, 0, K, ka, ar);

  f32x8 acc[2][2] = {};

  int kp0 = 0;
  for (int k0 = 0; k0 < K; k0 += 32, kp0 += 16) {
    // ---- commit staged A registers to LDS (bf16-packed) ----
    uint4 au;
    au.x = pack2bf(ar[0], ar[1]);
    au.y = pack2bf(ar[2], ar[3]);
    au.z = pack2bf(ar[4], ar[5]);
    au.w = pack2bf(ar[6], ar[7]);
    *(uint4*)&lA[ra * 20 + (ka >> 1)] = au;
    __syncthreads();

    // ---- prefetch next A slab (overlaps frag reads + WMMA) ----
    if (k0 + 32 < K)
      loadA(Arow, k0 + 32, K, ka, ar);

    // ---- B fragments straight from packed global weights ----
    Frag bfr[2];
    {
      const uint4* q0 = (const uint4*)(wrow0 + kp0);
      const uint4* q1 = (const uint4*)(wrow1 + kp0);
      bfr[0].q[0] = q0[0]; bfr[0].q[1] = q0[1];
      bfr[1].q[0] = q1[0]; bfr[1].q[1] = q1[1];
    }

    // ---- A fragments from LDS (contiguous k-pair map: kp = lhalf*8 + v) ----
    Frag af[2];
#pragma unroll
    for (int a = 0; a < 2; ++a)
#pragma unroll
      for (int v = 0; v < 8; ++v) {
        int kp = (lhalf << 3) + v;
        af[a].u[v] = lA[(wm * 32 + a * 16 + lrow) * 20 + kp];
      }

#pragma unroll
    for (int a = 0; a < 2; ++a)
#pragma unroll
      for (int b = 0; b < 2; ++b)
        acc[a][b] = __builtin_amdgcn_wmma_f32_16x16x32_bf16(
            false, af[a].v, false, bfr[b].v, (short)0, acc[a][b], false, false);

    __syncthreads();
  }

  // ---- epilogue: bias + activation (+ residual), row-guarded stores ----
#pragma unroll
  for (int a = 0; a < 2; ++a)
#pragma unroll
    for (int b = 0; b < 2; ++b) {
      int col = bn + wn * 32 + b * 16 + lrow;
      float bv = bias ? bias[col] : 0.0f;
#pragma unroll
      for (int v = 0; v < 8; ++v) {
        int row = bm + wm * 32 + a * 16 + v + 8 * lhalf;
        if (row >= M) continue;
        float val = acc[a][b][v] + bv;
        val = act_apply(val, act);
        if (resid) val += resid[(long long)row * N + col];
        C[(long long)row * N + col] = val;
      }
    }
}

// ---------------------------------------------------------------------------
// Small-N GEMM (N <= 4): one row per thread (att / coord2 / FC head 2).
// ---------------------------------------------------------------------------
__global__ void gemm_small_kernel(const float* __restrict__ A, const float* __restrict__ W,
                                  const float* __restrict__ bias, float* __restrict__ C,
                                  int M, int N, int K, int act)
{
  int r = blockIdx.x * blockDim.x + threadIdx.x;
  if (r >= M) return;
  const float* a = A + (long long)r * K;
  for (int n = 0; n < N; ++n) {
    float s = bias ? bias[n] : 0.0f;
    for (int k = 0; k < K; ++k) s += a[k] * W[k * N + n];
    C[(long long)r * N + n] = act_apply(s, act);
  }
}

// ---------------------------------------------------------------------------
// Graph kernels
// ---------------------------------------------------------------------------
__global__ void edge_count_kernel(const long long* __restrict__ row, float* __restrict__ cnt, int E)
{
  int e = blockIdx.x * blockDim.x + threadIdx.x;
  if (e < E) atomicAdd(&cnt[(int)row[e]], 1.0f);
}

__global__ void invcnt_kernel(float* __restrict__ cnt, int N)
{
  int i = blockIdx.x * blockDim.x + threadIdx.x;
  if (i < N) cnt[i] = 1.0f / fmaxf(cnt[i], 1.0f);
}

__global__ void edge_geom_kernel(const long long* __restrict__ row, const long long* __restrict__ col,
                                 const float* __restrict__ pos, float* __restrict__ diff,
                                 float* __restrict__ radial, int E)
{
  int e = blockIdx.x * blockDim.x + threadIdx.x;
  if (e >= E) return;
  int r = (int)row[e], c = (int)col[e];
  float dx = pos[r * 3 + 0] - pos[c * 3 + 0];
  float dy = pos[r * 3 + 1] - pos[c * 3 + 1];
  float dz = pos[r * 3 + 2] - pos[c * 3 + 2];
  diff[(long long)e * 3 + 0] = dx;
  diff[(long long)e * 3 + 1] = dy;
  diff[(long long)e * 3 + 2] = dz;
  radial[e] = dx * dx + dy * dy + dz * dz;
}

// concat row: [h[row] (256) | h[col] (256) | radial (1) | ef (128)] -> 641 cols
// output row stride 648 (16B-aligned rows for float4 GEMM staging)
#define CAT_K  641
#define CAT_LD 648
__global__ void build_concat_kernel(const long long* __restrict__ row, const long long* __restrict__ col,
                                    const float* __restrict__ h, const float* __restrict__ radial,
                                    const float* __restrict__ ef, float* __restrict__ out, int E)
{
  long long idx = (long long)blockIdx.x * blockDim.x + threadIdx.x;
  if (idx >= (long long)E * CAT_K) return;
  int e = (int)(idx / CAT_K);
  int j = (int)(idx % CAT_K);
  float v;
  if (j < 256)       v = h[(long long)row[e] * 256 + j];
  else if (j < 512)  v = h[(long long)col[e] * 256 + (j - 256)];
  else if (j == 512) v = radial[e];
  else               v = ef[(long long)e * 128 + (j - 513)];
  out[(long long)e * CAT_LD + j] = v;
}

__global__ void gate_kernel(float* __restrict__ m, const float* __restrict__ att, long long total)
{
  long long idx = (long long)blockIdx.x * blockDim.x + threadIdx.x;
  if (idx < total) m[idx] *= att[idx >> 8];   // att already sigmoid'ed, one per edge
}

__global__ void trans_atomic_kernel(const long long* __restrict__ row, const float* __restrict__ diff,
                                    const float* __restrict__ t, float* __restrict__ xacc, int E)
{
  int e = blockIdx.x * blockDim.x + threadIdx.x;
  if (e >= E) return;
  int r = (int)row[e];
  float tv = t[e];
  atomicAdd(&xacc[r * 3 + 0], diff[(long long)e * 3 + 0] * tv);
  atomicAdd(&xacc[r * 3 + 1], diff[(long long)e * 3 + 1] * tv);
  atomicAdd(&xacc[r * 3 + 2], diff[(long long)e * 3 + 2] * tv);
}

__global__ void coord_update_kernel(float* __restrict__ pos, const float* __restrict__ xacc,
                                    const float* __restrict__ inv, int N)
{
  int idx = blockIdx.x * blockDim.x + threadIdx.x;
  if (idx < N * 3) pos[idx] += xacc[idx] * inv[idx / 3];
}

__global__ void agg_atomic_kernel(const long long* __restrict__ row, const float* __restrict__ m,
                                  float* __restrict__ agg, int E)
{
  long long idx = (long long)blockIdx.x * blockDim.x + threadIdx.x;
  if (idx >= (long long)E * 256) return;
  int e = (int)(idx >> 8);
  int j = (int)(idx & 255);
  atomicAdd(&agg[(long long)row[e] * 256 + j], m[idx]);
}

__global__ void node_concat_kernel(const float* __restrict__ h, const float* __restrict__ agg,
                                   float* __restrict__ out, int N)
{
  long long idx = (long long)blockIdx.x * blockDim.x + threadIdx.x;
  if (idx >= (long long)N * 512) return;
  int n = (int)(idx >> 9);
  int j = (int)(idx & 511);
  out[idx] = (j < 256) ? h[(long long)n * 256 + j] : agg[(long long)n * 256 + (j - 256)];
}

// ---------------------------------------------------------------------------
// Host side
// ---------------------------------------------------------------------------
struct PW { const unsigned int* p; int kp2; };

static inline void gemm(const float* A, PW w, const float* bias, const float* resid,
                        float* C, int M, int N, int K, int lda, int act, hipStream_t s)
{
  dim3 g(N / 128, (M + 63) / 64);
  gemm_bf16_wmma_kernel<<<g, 256, 0, s>>>(A, w.p, bias, resid, C, M, N, K, lda, w.kp2, act);
}

static inline void smalln(const float* A, const float* W, const float* bias, float* C,
                          int M, int N, int K, int act, hipStream_t s)
{
  gemm_small_kernel<<<(M + 255) / 256, 256, 0, s>>>(A, W, bias, C, M, N, K, act);
}

struct EGCLp { const float *att_b, *att_w, *c1_b, *c1_w, *c2_w,
                           *e1_b, *e1_w, *e2_b, *e2_w, *n1_b, *n1_w, *n2_b, *n2_w;
               PW e1, e2, c1, n1, n2; };
struct EGp   { const float *embin_b, *embin_w, *embout_b, *embout_w;
               PW embin, embout; EGCLp L[4]; };
struct FCp   { const float *l1_b, *l1_w, *l2_b, *l2_w; PW l1; };

extern "C" void kernel_launch(void* const* d_in, const int* in_sizes, int n_in,
                              void* d_out, int out_size, void* d_ws, size_t ws_size,
                              hipStream_t stream)
{
  (void)n_in; (void)out_size; (void)ws_size;

  const float*     x_res = (const float*)d_in[0];
  const float*     x_pos = (const float*)d_in[1];
  const float*     e_ft  = (const float*)d_in[2];
  const long long* eidx  = (const long long*)d_in[3];

  const int E   = in_sizes[3] / 2;    // 320000
  const int Nn  = in_sizes[1] / 3;    // 10000
  const int INF = in_sizes[0] / Nn;   // 1024
  const long long* rowp = eidx;
  const long long* colp = eidx + E;

  // ---- unpack params (JAX pytree: dict keys sorted alphabetically) ----
  int pi = 4;
#define FP() ((const float*)d_in[pi++])
  const float* edge_fc_b = FP();
  const float* edge_fc_w = FP();
  EGp eg[4];
  for (int g = 0; g < 4; ++g) {
    eg[g].embin_b  = FP(); eg[g].embin_w  = FP();
    eg[g].embout_b = FP(); eg[g].embout_w = FP();
    for (int l = 0; l < 4; ++l) {               // sorted: att,coord1,coord2,edge1,edge2,node1,node2
      EGCLp& c = eg[g].L[l];
      c.att_b = FP(); c.att_w = FP();
      c.c1_b  = FP(); c.c1_w  = FP();
      c.c2_w  = FP();                            // coord2 has no bias
      c.e1_b  = FP(); c.e1_w  = FP();
      c.e2_b  = FP(); c.e2_w  = FP();
      c.n1_b  = FP(); c.n1_w  = FP();
      c.n2_b  = FP(); c.n2_w  = FP();
    }
  }
  FCp fc[4];
  for (int g = 0; g < 4; ++g) { fc[g].l1_b = FP(); fc[g].l1_w = FP(); fc[g].l2_b = FP(); fc[g].l2_w = FP(); }
#undef FP

  // ---- workspace bump allocator (256B aligned carve-outs) ----
  char* wsp = (char*)d_ws;
  auto alloc = [&](size_t bytes) -> char* {
    char* p = wsp;
    wsp += (bytes + 255) & ~(size_t)255;
    return p;
  };

  // pack a weight matrix (K,N fp32) -> bf16-pair tiles, once per launch
  auto packw = [&](const float* W, int K, int N) -> PW {
    int kp2 = ((K + 31) / 32) * 16;
    unsigned int* dst = (unsigned int*)alloc((size_t)N * kp2 * 4);
    long long tot = (long long)N * kp2;
    pack_weight_kernel<<<(unsigned)((tot + 255) / 256), 256, 0, stream>>>(W, dst, K, N, kp2);
    return PW{dst, kp2};
  };

  float* invc  = (float*)alloc((size_t)Nn * 4);
  float* efb   = (float*)alloc((size_t)E * 128 * 4);
  float* diffb = (float*)alloc((size_t)E * 3 * 4);
  float* radb  = (float*)alloc((size_t)E * 4);
  float* catE  = (float*)alloc((size_t)E * CAT_LD * 4);
  float* m1    = (float*)alloc((size_t)E * 256 * 4);
  float* m2    = (float*)alloc((size_t)E * 256 * 4);
  float* attb  = (float*)alloc((size_t)E * 4);
  float* tb    = (float*)alloc((size_t)E * 4);
  float* xacc  = (float*)alloc((size_t)Nn * 3 * 4);
  float* agg   = (float*)alloc((size_t)Nn * 256 * 4);
  float* ncat  = (float*)alloc((size_t)Nn * 512 * 4);
  float* ntmp  = (float*)alloc((size_t)Nn * 256 * 4);
  float* hhid  = (float*)alloc((size_t)Nn * 256 * 4);
  float* pbuf  = (float*)alloc((size_t)Nn * 3 * 4);
  float* fctmp = (float*)alloc((size_t)Nn * 128 * 4);

  int outf[4] = { 512, 512, 256, 128 };
  int inff[4] = { INF, 512, 512, 256 };

  // ---- pre-pack all WMMA weights (few MB total; L2-resident afterwards) ----
  PW pw_edge_fc = packw(edge_fc_w, 16, 128);
  for (int g = 0; g < 4; ++g) {
    eg[g].embin  = packw(eg[g].embin_w, inff[g], 256);
    eg[g].embout = packw(eg[g].embout_w, 256, outf[g]);
    for (int l = 0; l < 4; ++l) {
      EGCLp& c = eg[g].L[l];
      c.e1 = packw(c.e1_w, CAT_K, 256);
      c.e2 = packw(c.e2_w, 256, 256);
      c.c1 = packw(c.c1_w, 256, 256);
      c.n1 = packw(c.n1_w, 512, 256);
      c.n2 = packw(c.n2_w, 256, 256);
    }
    fc[g].l1 = packw(fc[g].l1_w, outf[g], 128);
  }

  float* out = (float*)d_out;
  // d_out layout: out4,out3,out2,out1,h4,h3,h2,h1
  size_t o_out4 = 0;
  size_t o_out3 = o_out4 + (size_t)Nn * 2;
  size_t o_out2 = o_out3 + (size_t)Nn * 2;
  size_t o_out1 = o_out2 + (size_t)Nn * 2;
  size_t o_h4   = o_out1 + (size_t)Nn * 2;
  size_t o_h3   = o_h4 + (size_t)Nn * 128;
  size_t o_h2   = o_h3 + (size_t)Nn * 256;
  size_t o_h1   = o_h2 + (size_t)Nn * 512;
  float* hout[4]  = { out + o_h1, out + o_h2, out + o_h3, out + o_h4 };
  float* fcout[4] = { out + o_out1, out + o_out2, out + o_out3, out + o_out4 };

  const int TB = 256;
  // ---- per-node inverse edge count (constant across layers) ----
  hipMemsetAsync(invc, 0, (size_t)Nn * 4, stream);
  edge_count_kernel<<<(E + TB - 1) / TB, TB, 0, stream>>>(rowp, invc, E);
  invcnt_kernel<<<(Nn + TB - 1) / TB, TB, 0, stream>>>(invc, Nn);

  // ---- edge feature embedding: ef = edge_feat @ W + b  [E,16]x[16,128] ----
  gemm(e_ft, pw_edge_fc, edge_fc_b, nullptr, efb, E, 128, 16, 16, ACT_NONE, stream);

  // ---- coords working copy (do not mutate inputs) ----
  hipMemcpyAsync(pbuf, x_pos, (size_t)Nn * 3 * 4, hipMemcpyDeviceToDevice, stream);

  const float* hin = x_res;
  long long ecat = (long long)E * CAT_K;
  long long e256 = (long long)E * 256;
  long long n512 = (long long)Nn * 512;

  for (int g = 0; g < 4; ++g) {
    // emb_in
    gemm(hin, eg[g].embin, eg[g].embin_b, nullptr, hhid, Nn, 256, inff[g], inff[g], ACT_NONE, stream);

    for (int l = 0; l < 4; ++l) {
      const EGCLp& c = eg[g].L[l];
      edge_geom_kernel<<<(E + TB - 1) / TB, TB, 0, stream>>>(rowp, colp, pbuf, diffb, radb, E);
      build_concat_kernel<<<(unsigned)((ecat + TB - 1) / TB), TB, 0, stream>>>(
          rowp, colp, hhid, radb, efb, catE, E);

      // edge model: m = silu(edge2(silu(edge1(cat))))
      gemm(catE, c.e1, c.e1_b, nullptr, m1, E, 256, CAT_K, CAT_LD, ACT_SILU, stream);
      gemm(m1,   c.e2, c.e2_b, nullptr, m2, E, 256, 256, 256, ACT_SILU, stream);
      // attention gate
      smalln(m2, c.att_w, c.att_b, attb, E, 1, 256, ACT_SIGMOID, stream);
      gate_kernel<<<(unsigned)((e256 + TB - 1) / TB), TB, 0, stream>>>(m2, attb, e256);

      // coord model: t = tanh(coord2(silu(coord1(m))))
      gemm(m2, c.c1, c.c1_b, nullptr, m1, E, 256, 256, 256, ACT_SILU, stream);
      smalln(m1, c.c2_w, nullptr, tb, E, 1, 256, ACT_TANH, stream);
      hipMemsetAsync(xacc, 0, (size_t)Nn * 3 * 4, stream);
      trans_atomic_kernel<<<(E + TB - 1) / TB, TB, 0, stream>>>(rowp, diffb, tb, xacc, E);
      coord_update_kernel<<<(Nn * 3 + TB - 1) / TB, TB, 0, stream>>>(pbuf, xacc, invc, Nn);

      // node model: h += node2(silu(node1([h | segsum(m)])))
      hipMemsetAsync(agg, 0, (size_t)Nn * 256 * 4, stream);
      agg_atomic_kernel<<<(unsigned)((e256 + TB - 1) / TB), TB, 0, stream>>>(rowp, m2, agg, E);
      node_concat_kernel<<<(unsigned)((n512 + TB - 1) / TB), TB, 0, stream>>>(hhid, agg, ncat, Nn);
      gemm(ncat, c.n1, c.n1_b, nullptr, ntmp, Nn, 256, 512, 512, ACT_SILU, stream);
      gemm(ntmp, c.n2, c.n2_b, hhid, hhid, Nn, 256, 256, 256, ACT_NONE, stream);  // residual
    }
    // emb_out -> h_g straight into d_out
    gemm(hhid, eg[g].embout, eg[g].embout_b, nullptr, hout[g], Nn, outf[g], 256, 256, ACT_NONE, stream);
    hin = hout[g];
  }

  // ---- FC heads: out_g = lin2(relu(lin1(h_g))) ----
  for (int g = 0; g < 4; ++g) {
    gemm(hout[g], fc[g].l1, fc[g].l1_b, nullptr, fctmp, Nn, 128, outf[g], outf[g], ACT_RELU, stream);
    smalln(fctmp, fc[g].l2_w, fc[g].l2_b, fcout[g], Nn, 2, 128, ACT_NONE, stream);
  }
}
